// KANHeadSpline_6837587935648
// MI455X (gfx1250) — matrix operand order repaired and checked by reading, not compiled, not run on hardware
//
#include <hip/hip_runtime.h>

typedef __attribute__((ext_vector_type(16))) _Float16 v16h;
typedef __attribute__((ext_vector_type(2)))  _Float16 v2h;
typedef __attribute__((ext_vector_type(8)))  float    v8f;

#define HW_    (192*192)
#define CIN    64
#define HID    64
#define OUTC   4
#define NKNOT  16
#define XMIN_  (-2.0f)
#define DELTA_ (4.0f/15.0f)
#define XPAD   72   // padded row stride (halves): 144B = 36 words, conflict-free A-frag reads

#if defined(__has_builtin)
#if __has_builtin(__builtin_amdgcn_permlane16)
#define HAVE_PERMLANE16 1
#endif
#endif

// One xor-butterfly step within each 16-lane row, done in VALU via
// v_permlane16_b32 (nibble i of {lo,hi} selects source lane i^mask).
__device__ __forceinline__ float bfly16(float v, unsigned lo, unsigned hi, int mask) {
#ifdef HAVE_PERMLANE16
  unsigned u = __float_as_uint(v);
  unsigned r = __builtin_amdgcn_permlane16(u, u, lo, hi, false, false);
  return v + __uint_as_float(r);
#else
  return v + __shfl_xor(v, mask, 32);
#endif
}

// Full 16-lane sum broadcast (masks 1,2,4,8 — stays inside each lane half).
__device__ __forceinline__ float red16(float v) {
  v = bfly16(v, 0x67452301u, 0xEFCDAB89u, 1);
  v = bfly16(v, 0x54761032u, 0xDCFE98BAu, 2);
  v = bfly16(v, 0x32107654u, 0xBA98FEDCu, 4);
  v = bfly16(v, 0xFEDCBA98u, 0x76543210u, 8);
  return v;
}

// One 16x64 * 64x64 GEMM using 8 x v_wmma_f32_16x16x32_f16.
__device__ __forceinline__ void gemm64(const _Float16 (*sA)[XPAD],
                                       const _Float16 (*Bf)[4][32][16],
                                       int lane, v8f acc[4]) {
  const int M    = lane & 15;
  const int kadd = (lane < 16) ? 0 : 8;   // 16-bit A layout: K 0-7 | 8-15 halves
#pragma unroll
  for (int kb = 0; kb < 2; ++kb) {
    v16h a;
#pragma unroll
    for (int j = 0; j < 16; ++j) {
      // a[j] holds K = kb*32 + (j/8)*16 + kadd + (j%8)  (ISA 7.12.2, 16-bit A 16x32)
      a[j] = sA[M][kb * 32 + ((j >> 3) << 4) + kadd + (j & 7)];
    }
#pragma unroll
    for (int nt = 0; nt < 4; ++nt) {
      v16h b = *(const v16h*)(&Bf[kb][nt][lane][0]);
      acc[nt] = __builtin_amdgcn_wmma_f32_16x16x32_f16(
          /*neg_a=*/false, a, /*neg_b=*/false, b,
          /*c_mod=*/(short)0, acc[nt], /*reuse_a=*/false, /*reuse_b=*/false);
    }
  }
}

__global__ __launch_bounds__(256)
void kan_head_fused(const float* __restrict__ x,
                    const float* __restrict__ Win,
                    const float* __restrict__ bin,
                    const float* __restrict__ gam,
                    const float* __restrict__ bet,
                    const float* __restrict__ splW,
                    const float* __restrict__ splB,
                    const float* __restrict__ Wmix,
                    const float* __restrict__ bmix,
                    const float* __restrict__ Wout,
                    const float* __restrict__ bout,
                    float* __restrict__ out) {
  // B-fragment-swizzled weights (f16), per ISA B layout:
  // lane = column N (lane%16), lanes 0-15 hold K 0-7,16-23, lanes 16-31 hold K 8-15,24-31
  __shared__ __attribute__((aligned(32))) _Float16 sWinF[2][4][32][16];   // 8 KB
  __shared__ __attribute__((aligned(32))) _Float16 sWmixF[2][4][32][16];  // 8 KB
  __shared__ __attribute__((aligned(16))) _Float16 sX[8][16][XPAD];       // 18 KB (X tile, reused for Z)
  __shared__ float sSplW[HID][NKNOT];                                     // 4 KB
  __shared__ float sGam[HID], sBet[HID], sBmix[HID], sSplB[HID], sBin[HID];
  __shared__ float sWo[HID][OUTC];
  __shared__ float sBo[OUTC];

  const int tid   = threadIdx.x;
  const int w     = tid >> 5;
  const int lane  = tid & 31;
  const int pbase = blockIdx.x * 128 + w * 16;       // HW_ % 128 == 0 -> one batch idx
  const int bidx  = pbase / HW_;
  const int s0    = pbase - bidx * HW_;
  const float* xb = x + (size_t)bidx * CIN * HW_ + s0;

  // Prefetch this wave's X tile (64 cachelines of 64B) so HBM latency
  // overlaps the per-block weight staging below.
  __builtin_prefetch(xb + (size_t)lane * HW_, 0, 0);
  __builtin_prefetch(xb + (size_t)(lane + 32) * HW_, 0, 0);

  // ---- Phase 0: stage & swizzle weights (once per block) ----
  for (int e = tid; e < 2 * 4 * 32 * 16; e += 256) {
    int kb = e >> 11;
    int r  = e & 2047;
    int nt = r >> 9;  r &= 511;
    int ln = r >> 4;
    int j  = r & 15;
    int c  = kb * 32 + ((j >> 3) << 4) + ((ln < 16) ? 0 : 8) + (j & 7);  // K (input channel)
    int h  = nt * 16 + (ln & 15);                 // N (output column)
    sWinF [kb][nt][ln][j] = (_Float16)Win [c * HID + h];
    sWmixF[kb][nt][ln][j] = (_Float16)Wmix[c * HID + h];
  }
  for (int e = tid; e < HID * NKNOT; e += 256) sSplW[e >> 4][e & 15] = splW[e];
  for (int e = tid; e < HID * OUTC;  e += 256) sWo[e >> 2][e & 3]    = Wout[e];
  if (tid < HID) {
    sGam[tid]  = gam[tid];  sBet[tid]  = bet[tid];
    sBmix[tid] = bmix[tid]; sSplB[tid] = splB[tid]; sBin[tid] = bin[tid];
  }
  if (tid < OUTC) sBo[tid] = bout[tid];
  __syncthreads();

  // ---- Phase 1: stage 16-pixel x 64-channel X tile as f16 ----
  // Paired float4 loads (channels c, c+1; 4 contiguous pixels) -> packed
  // v2h stores (one ds_store_b32 each). 8 global_load_b128 per lane total.
#pragma unroll
  for (int t = 0; t < 4; ++t) {
    int q  = t * 32 + lane;       // 0..127
    int c  = (q >> 2) * 2;        // even channel
    int p4 = (q & 3) * 4;         // pixel group
    float4 a = *(const float4*)(xb + (size_t)c * HW_ + p4);
    float4 b = *(const float4*)(xb + (size_t)(c + 1) * HW_ + p4);
    *(v2h*)&sX[w][p4 + 0][c] = (v2h){(_Float16)a.x, (_Float16)b.x};
    *(v2h*)&sX[w][p4 + 1][c] = (v2h){(_Float16)a.y, (_Float16)b.y};
    *(v2h*)&sX[w][p4 + 2][c] = (v2h){(_Float16)a.z, (_Float16)b.z};
    *(v2h*)&sX[w][p4 + 3][c] = (v2h){(_Float16)a.w, (_Float16)b.w};
  }

  // ---- Phase 2: GEMM1  Z(16x64) = X(16x64) @ W_in ----
  v8f acc[4] = {};
  gemm64(sX[w], sWinF, lane, acc);

  // ---- Phase 3: +b_in, LayerNorm (row-wise over 64), tanh, hat-spline, SiLU ----
  // D layout: acc[nt][r] = z[M = r + 8*(lane>=16)][h = nt*16 + lane%16]
  const int rowoff = (lane < 16) ? 0 : 8;
  const int col    = lane & 15;
#pragma unroll
  for (int r = 0; r < 8; ++r) {
    float s1 = 0.f, s2 = 0.f;
#pragma unroll
    for (int nt = 0; nt < 4; ++nt) {
      float v = acc[nt][r] + sBin[nt * 16 + col];
      acc[nt][r] = v;
      s1 += v; s2 += v * v;
    }
    s1 = red16(s1);
    s2 = red16(s2);
    float mean = s1 * (1.f / 64.f);
    float var  = s2 * (1.f / 64.f) - mean * mean;
    float rinv = rsqrtf(var + 1e-5f);
#pragma unroll
    for (int nt = 0; nt < 4; ++nt) {
      int h   = nt * 16 + col;
      float v = (acc[nt][r] - mean) * rinv * sGam[h] + sBet[h];
      v = tanhf(v);
      // hat-spline: only 2 adjacent knots active -> gather + lerp
      float t  = (v - XMIN_) * (1.0f / DELTA_);
      int   i0 = (int)floorf(t);
      i0 = min(max(i0, 0), NKNOT - 2);
      float f  = t - (float)i0;
      float sv = sSplW[h][i0] * (1.f - f) + sSplW[h][i0 + 1] * f + sSplB[h];
      sv = sv / (1.f + __expf(-sv));               // SiLU
      sX[w][r + rowoff][h] = (_Float16)sv;         // same-wave LDS: ordered, no barrier
    }
  }

  // ---- Phase 4: GEMM2  Y = SiLU(Z) @ W_mix ----
  v8f acc2[4] = {};
  gemm64(sX[w], sWmixF, lane, acc2);

  // ---- Phase 5: +b_mix, SiLU, 64->4 projection via FMAs + butterfly, store ----
  const size_t obase = (size_t)bidx * OUTC * HW_ + s0;
#pragma unroll
  for (int r = 0; r < 8; ++r) {
    float po[OUTC] = {0.f, 0.f, 0.f, 0.f};
#pragma unroll
    for (int nt = 0; nt < 4; ++nt) {
      int h   = nt * 16 + col;
      float y = acc2[nt][r] + sBmix[h];
      y = y / (1.f + __expf(-y));
#pragma unroll
      for (int o = 0; o < OUTC; ++o) po[o] += y * sWo[h][o];
    }
#pragma unroll
    for (int o = 0; o < OUTC; ++o) po[o] = red16(po[o]);
    if (col < OUTC) {                              // lanes 0-3 / 16-19 write
      int Mrow = r + rowoff;
      out[obase + (size_t)col * HW_ + Mrow] = po[col] + sBo[col];
    }
  }
}

extern "C" void kernel_launch(void* const* d_in, const int* in_sizes, int n_in,
                              void* d_out, int out_size, void* d_ws, size_t ws_size,
                              hipStream_t stream) {
  const float* x    = (const float*)d_in[0];
  const float* Win  = (const float*)d_in[1];
  const float* bin  = (const float*)d_in[2];
  const float* gam  = (const float*)d_in[3];
  const float* bet  = (const float*)d_in[4];
  const float* splW = (const float*)d_in[5];
  const float* splB = (const float*)d_in[6];
  const float* Wmix = (const float*)d_in[7];
  const float* bmix = (const float*)d_in[8];
  const float* Wout = (const float*)d_in[9];
  const float* bout = (const float*)d_in[10];
  float* out = (float*)d_out;

  const int npix   = in_sizes[0] / CIN;  // 147456
  const int blocks = npix / 128;         // 1152
  kan_head_fused<<<blocks, 256, 0, stream>>>(x, Win, bin, gam, bet, splW, splB,
                                             Wmix, bmix, Wout, bout, out);
}
